// Attention_70566312673776
// MI455X (gfx1250) — matrix operand order repaired and checked by reading, not compiled, not run on hardware
//
#include <hip/hip_runtime.h>
#include <hip/hip_bf16.h>
#include <math.h>

// ---------------------------------------------------------------------------
// Types for CDNA5 WMMA (wave32): D = A(16x32 bf16) * B(32x16 bf16) + C(f32)
// ---------------------------------------------------------------------------
typedef __bf16 bf16_t;
typedef __attribute__((ext_vector_type(16))) __bf16 v16bf;
typedef __attribute__((ext_vector_type(8)))  float  v8f;
typedef int vi4 __attribute__((vector_size(16)));   // matches builtin V4i

union Frag16 {           // 16 bf16 = 32 bytes = two 128-bit loads
    v16bf v;
    uint4 q[2];
};

__device__ __forceinline__ v8f wmma_bf16(const v16bf& a, const v16bf& b, const v8f& c) {
    return __builtin_amdgcn_wmma_f32_16x16x32_bf16(false, a, false, b,
                                                   (short)0, c, false, false);
}

__device__ __forceinline__ v8f v8f_zero() {
    v8f z;
#pragma unroll
    for (int i = 0; i < 8; ++i) z[i] = 0.0f;
    return z;
}

// ---------------------------------------------------------------------------
// CDNA5 async global->LDS copy (ASYNCcnt-tracked) with sync fallback.
// Builtin signature: (v4i addrspace(1)* src, v4i addrspace(3)* dst,
//                     imm offset, imm cpol)
// ---------------------------------------------------------------------------
#if defined(__has_builtin)
#if __has_builtin(__builtin_amdgcn_global_load_async_to_lds_b128)
#define HAVE_ASYNC_LDS 1
#endif
#endif

__device__ __forceinline__ void cp16_g2l(void* lds_dst, const void* g_src) {
#ifdef HAVE_ASYNC_LDS
    __builtin_amdgcn_global_load_async_to_lds_b128(
        (__attribute__((address_space(1))) vi4*)(g_src),
        (__attribute__((address_space(3))) vi4*)(lds_dst),
        0, 0);
#else
    *(uint4*)lds_dst = *(const uint4*)g_src;
#endif
}

__device__ __forceinline__ void wait_async_all() {
#ifdef HAVE_ASYNC_LDS
#if __has_builtin(__builtin_amdgcn_s_wait_asynccnt)
    __builtin_amdgcn_s_wait_asynccnt(0);
#else
    asm volatile("s_wait_asynccnt 0" ::: "memory");
#endif
#endif
}

// ---------------------------------------------------------------------------
// fp32 -> bf16 conversion
// ---------------------------------------------------------------------------
__global__ void f32_to_bf16_kernel(const float* __restrict__ in,
                                   bf16_t* __restrict__ out, int n) {
    int i = blockIdx.x * blockDim.x + threadIdx.x;
    if (i < n) out[i] = (bf16_t)in[i];
}

// ---------------------------------------------------------------------------
// GEMM:  C[M,N] = A[M,K] * W[N,K]^T   (A,W bf16 row-major; C f32 or bf16)
// Workgroup = 256 threads = 8 waves; block tile 128x128, wave tile 64x32.
// A/B tiles (128x64 each) double-buffered in LDS via async global->LDS.
// ---------------------------------------------------------------------------
#define GPAD 72   // LDS row stride for 64-wide k-slab (bf16), 16B aligned

template <bool OUT_F32>
__global__ __launch_bounds__(256) void gemm_nt_kernel(
        const bf16_t* __restrict__ A, const bf16_t* __restrict__ W,
        void* __restrict__ out, int M, int N, int K) {
    __shared__ __align__(16) bf16_t sA[2][128 * GPAD];
    __shared__ __align__(16) bf16_t sB[2][128 * GPAD];

    const int lane = threadIdx.x & 31;
    const int wave = threadIdx.x >> 5;
    const int half = lane >> 4;
    const int l16  = lane & 15;

    const int Mblock = blockIdx.y * 128;
    const int Nblock = blockIdx.x * 128;
    const int mloc   = (wave & 1) * 64;      // wave's M offset in tile
    const int nloc   = (wave >> 1) * 32;     // wave's N offset in tile

    v8f acc[4][2];
#pragma unroll
    for (int mi = 0; mi < 4; ++mi)
#pragma unroll
        for (int ni = 0; ni < 2; ++ni) acc[mi][ni] = v8f_zero();

    // staging map: 512 chunks of 16B per 128x64 tile -> 2 chunks per thread
    auto stage = [&](int buf, int kk) {
#pragma unroll
        for (int c = 0; c < 2; ++c) {
            int id  = threadIdx.x + c * 256;   // 0..511
            int row = id >> 2;
            int seg = id & 3;
            cp16_g2l(&sA[buf][row * GPAD + seg * 16],
                     A + (size_t)(Mblock + row) * K + kk + seg * 16);
            cp16_g2l(&sB[buf][row * GPAD + seg * 16],
                     W + (size_t)(Nblock + row) * K + kk + seg * 16);
        }
    };

    stage(0, 0);
    wait_async_all();
    __syncthreads();

    int buf = 0;
    for (int kk = 0; kk < K; kk += 64) {
        if (kk + 64 < K) stage(buf ^ 1, kk + 64);

#pragma unroll
        for (int kc = 0; kc < 64; kc += 32) {
            Frag16 af[4];
#pragma unroll
            for (int mi = 0; mi < 4; ++mi) {
                const bf16_t* p = &sA[buf][(mloc + mi * 16 + l16) * GPAD + kc + 8 * half];
                af[mi].q[0] = *(const uint4*)p;
                af[mi].q[1] = *(const uint4*)(p + 16);
            }
            Frag16 bfr[2];
#pragma unroll
            for (int ni = 0; ni < 2; ++ni) {
                const bf16_t* p = &sB[buf][(nloc + ni * 16 + l16) * GPAD + kc + 8 * half];
                bfr[ni].q[0] = *(const uint4*)p;
                bfr[ni].q[1] = *(const uint4*)(p + 16);
            }
#pragma unroll
            for (int mi = 0; mi < 4; ++mi)
#pragma unroll
                for (int ni = 0; ni < 2; ++ni)
                    acc[mi][ni] = wmma_bf16(af[mi].v, bfr[ni].v, acc[mi][ni]);
        }

        wait_async_all();
        __syncthreads();
        buf ^= 1;
    }

#pragma unroll
    for (int mi = 0; mi < 4; ++mi)
#pragma unroll
        for (int ni = 0; ni < 2; ++ni)
#pragma unroll
            for (int r = 0; r < 8; ++r) {
                int row = Mblock + mloc + mi * 16 + r + 8 * half;
                int col = Nblock + nloc + ni * 16 + l16;
                if (OUT_F32)
                    ((float*)out)[(size_t)row * N + col] = acc[mi][ni][r];
                else
                    ((bf16_t*)out)[(size_t)row * N + col] = (bf16_t)acc[mi][ni][r];
            }
}

// ---------------------------------------------------------------------------
// RoPE (interleaved pairs) in-place on bf16 Q or K; 1/sqrt(D) folded into Q.
// ---------------------------------------------------------------------------
__global__ void rope_kernel(bf16_t* __restrict__ buf, float scale, int total) {
    int tid = blockIdx.x * blockDim.x + threadIdx.x;
    if (tid >= total) return;
    int i = tid & 63;
    int h = (tid >> 6) & 15;
    int s = (tid >> 10) & 2047;
    int b = tid >> 21;
    float inv = __powf(10000.0f, -(float)i * (1.0f / 64.0f));
    float ang = (float)s * inv;
    float c  = __cosf(ang);
    float sn = __sinf(ang);
    size_t off = ((size_t)(b * 2048 + s)) * 2048 + h * 128 + 2 * i;
    float xe = (float)buf[off];
    float xo = (float)buf[off + 1];
    buf[off]     = (bf16_t)((xe * c - xo * sn) * scale);
    buf[off + 1] = (bf16_t)((xe * sn + xo * c) * scale);
}

// ---------------------------------------------------------------------------
// Flash attention (causal, online softmax), bf16 QK^T and PV via WMMA.
// Grid: (S/128, H, B). Block: 256 threads = 8 waves; wave owns 16 query rows.
// ---------------------------------------------------------------------------
#define KPAD 136   // K tile: 32 x 136 bf16
#define VPAD 40    // V^T:   128 x 40 bf16
#define PPAD 40    // per-wave P tile: 16 x 40 bf16

__global__ __launch_bounds__(256) void attn_kernel(
        const bf16_t* __restrict__ Q, const bf16_t* __restrict__ Kb,
        const bf16_t* __restrict__ Vb, bf16_t* __restrict__ O, int S) {
    __shared__ __align__(16) bf16_t sK[32 * KPAD];
    __shared__ __align__(16) bf16_t sVt[128 * VPAD];
    __shared__ __align__(16) bf16_t sP[8][16 * PPAD];

    const int lane = threadIdx.x & 31;
    const int wave = threadIdx.x >> 5;
    const int half = lane >> 4;
    const int l16  = lane & 15;

    const int b = blockIdx.z;
    const int h = blockIdx.y;
    const size_t base = ((size_t)b * S) * 2048 + (size_t)h * 128;

    const int qblock = blockIdx.x * 128;
    const int qbase  = qblock + wave * 16;

    // Q fragments (D = 128 -> 4 chunks of K=32), pre-scaled by 1/sqrt(D)
    Frag16 qf[4];
#pragma unroll
    for (int c4 = 0; c4 < 4; ++c4) {
        const bf16_t* p = Q + base + (size_t)(qbase + l16) * 2048 + c4 * 32 + 8 * half;
        qf[c4].q[0] = *(const uint4*)p;
        qf[c4].q[1] = *(const uint4*)(p + 16);
    }

    float m_run[8], l_run[8];
    v8f acc[8];
#pragma unroll
    for (int r = 0; r < 8; ++r) { m_run[r] = -__builtin_inff(); l_run[r] = 0.0f; }
#pragma unroll
    for (int j = 0; j < 8; ++j) acc[j] = v8f_zero();

    const int kend = qblock + 128;           // uniform across workgroup
    const int ldr  = threadIdx.x >> 3;       // 0..31 : key row within tile
    const int seg8 = threadIdx.x & 7;        // 0..7  : 16-element segment of D

    for (int kb = 0; kb < kend; kb += 32) {
        __syncthreads();
        {   // stage K tile 32x128 (row-major, padded) via async global->LDS
            const bf16_t* src = Kb + base + (size_t)(kb + ldr) * 2048 + seg8 * 16;
            cp16_g2l(&sK[ldr * KPAD + seg8 * 16],     src);
            cp16_g2l(&sK[ldr * KPAD + seg8 * 16 + 8], src + 8);
        }
        {   // stage V tile transposed: sVt[d][key]
            const bf16_t* src = Vb + base + (size_t)(kb + ldr) * 2048 + seg8 * 16;
            Frag16 t;
            t.q[0] = ((const uint4*)src)[0];
            t.q[1] = ((const uint4*)src)[1];
#pragma unroll
            for (int i = 0; i < 16; ++i)
                sVt[(seg8 * 16 + i) * VPAD + ldr] = t.v[i];
        }
        wait_async_all();
        __syncthreads();

        if (kb <= qbase + 15) {              // wave-uniform causal skip
            v8f s0 = v8f_zero(), s1 = v8f_zero();
#pragma unroll
            for (int c4 = 0; c4 < 4; ++c4) {
                Frag16 k0, k1;
                const bf16_t* p0 = &sK[(0 * 16 + l16) * KPAD + c4 * 32 + 8 * half];
                const bf16_t* p1 = &sK[(1 * 16 + l16) * KPAD + c4 * 32 + 8 * half];
                k0.q[0] = *(const uint4*)p0; k0.q[1] = *(const uint4*)(p0 + 16);
                k1.q[0] = *(const uint4*)p1; k1.q[1] = *(const uint4*)(p1 + 16);
                s0 = wmma_bf16(qf[c4].v, k0.v, s0);
                s1 = wmma_bf16(qf[c4].v, k1.v, s1);
            }
            if (kb + 31 > qbase) {           // causal mask on edge blocks
#pragma unroll
                for (int r = 0; r < 8; ++r) {
                    int qrow = qbase + r + 8 * half;
                    if (kb + l16      > qrow) s0[r] = -__builtin_inff();
                    if (kb + 16 + l16 > qrow) s1[r] = -__builtin_inff();
                }
            }
            // online softmax: each row lives across the 16 lanes of a half
#pragma unroll
            for (int r = 0; r < 8; ++r) {
                float mx = fmaxf(s0[r], s1[r]);
#pragma unroll
                for (int off = 1; off < 16; off <<= 1)
                    mx = fmaxf(mx, __shfl_xor(mx, off));
                float mn    = fmaxf(m_run[r], mx);
                float alpha = __expf(m_run[r] - mn);
                float p0    = __expf(s0[r] - mn);
                float p1    = __expf(s1[r] - mn);
                float sum   = p0 + p1;
#pragma unroll
                for (int off = 1; off < 16; off <<= 1)
                    sum += __shfl_xor(sum, off);
                l_run[r] = l_run[r] * alpha + sum;
                m_run[r] = mn;
                s0[r] = p0;
                s1[r] = p1;
#pragma unroll
                for (int j = 0; j < 8; ++j) acc[j][r] *= alpha;
            }
            // redistribute P (C layout) -> A layout via per-wave LDS tile
#pragma unroll
            for (int r = 0; r < 8; ++r) {
                int m = r + 8 * half;
                sP[wave][m * PPAD + l16]      = (bf16_t)s0[r];
                sP[wave][m * PPAD + 16 + l16] = (bf16_t)s1[r];
            }
            asm volatile("s_wait_dscnt 0" ::: "memory");
            Frag16 pf;
            const bf16_t* pp = &sP[wave][l16 * PPAD + 8 * half];
            pf.q[0] = *(const uint4*)pp;
            pf.q[1] = *(const uint4*)(pp + 16);
#pragma unroll
            for (int j = 0; j < 8; ++j) {
                Frag16 vf;
                const bf16_t* vp = &sVt[(j * 16 + l16) * VPAD + 8 * half];
                vf.q[0] = *(const uint4*)vp;
                vf.q[1] = *(const uint4*)(vp + 16);
                acc[j] = wmma_bf16(pf.v, vf.v, acc[j]);
            }
        }
    }

    // epilogue: normalize and store bf16 in (B,S,H*D) layout
#pragma unroll
    for (int j = 0; j < 8; ++j)
#pragma unroll
        for (int r = 0; r < 8; ++r) {
            int row = qbase + r + 8 * half;
            int col = j * 16 + l16;
            O[base + (size_t)row * 2048 + col] =
                (bf16_t)(acc[j][r] / l_run[r]);
        }
}

// ---------------------------------------------------------------------------
// Host-side orchestration
// ---------------------------------------------------------------------------
extern "C" void kernel_launch(void* const* d_in, const int* in_sizes, int n_in,
                              void* d_out, int out_size, void* d_ws, size_t ws_size,
                              hipStream_t stream) {
    (void)in_sizes; (void)n_in; (void)out_size; (void)ws_size;
    const int B = 2, S = 2048, DIM = 2048;
    const int M = B * S;               // 4096
    const size_t MN = (size_t)M * DIM;
    const size_t NN = (size_t)DIM * DIM;

    const float* x  = (const float*)d_in[0];
    const float* wq = (const float*)d_in[1];
    const float* wk = (const float*)d_in[2];
    const float* wv = (const float*)d_in[3];
    const float* wo = (const float*)d_in[4];

    char* p = (char*)d_ws;
    bf16_t* xb  = (bf16_t*)p; p += MN * sizeof(bf16_t);
    bf16_t* wqb = (bf16_t*)p; p += NN * sizeof(bf16_t);
    bf16_t* wkb = (bf16_t*)p; p += NN * sizeof(bf16_t);
    bf16_t* wvb = (bf16_t*)p; p += NN * sizeof(bf16_t);
    bf16_t* wob = (bf16_t*)p; p += NN * sizeof(bf16_t);
    bf16_t* Qb  = (bf16_t*)p; p += MN * sizeof(bf16_t);
    bf16_t* Kb  = (bf16_t*)p; p += MN * sizeof(bf16_t);
    bf16_t* Vb  = (bf16_t*)p; p += MN * sizeof(bf16_t);
    bf16_t* Ab  = (bf16_t*)p; p += MN * sizeof(bf16_t);

    auto cvt = [&](const float* src, bf16_t* dst, size_t n) {
        f32_to_bf16_kernel<<<(unsigned)((n + 255) / 256), 256, 0, stream>>>(src, dst, (int)n);
    };
    cvt(x, xb, MN);
    cvt(wq, wqb, NN);
    cvt(wk, wkb, NN);
    cvt(wv, wvb, NN);
    cvt(wo, wob, NN);

    dim3 ggrid(DIM / 128, M / 128);    // (16, 32)
    gemm_nt_kernel<false><<<ggrid, 256, 0, stream>>>(xb, wqb, Qb, M, DIM, DIM);
    gemm_nt_kernel<false><<<ggrid, 256, 0, stream>>>(xb, wkb, Kb, M, DIM, DIM);
    gemm_nt_kernel<false><<<ggrid, 256, 0, stream>>>(xb, wvb, Vb, M, DIM, DIM);

    const int rope_total = B * S * 16 * 64;
    const float qscale = 0.08838834764831845f; // 1/sqrt(128)
    rope_kernel<<<rope_total / 256, 256, 0, stream>>>(Qb, qscale, rope_total);
    rope_kernel<<<rope_total / 256, 256, 0, stream>>>(Kb, 1.0f, rope_total);

    attn_kernel<<<dim3(S / 128, 16, B), 256, 0, stream>>>(Qb, Kb, Vb, Ab, S);

    gemm_nt_kernel<true><<<ggrid, 256, 0, stream>>>(Ab, wob, (float*)d_out, M, DIM, DIM);
}
